// SteeredConv2d_78769700209279
// MI455X (gfx1250) — matrix operand order repaired and checked by reading, not compiled, not run on hardware
//
#include <hip/hip_runtime.h>
#include <math.h>

// ---------------------------------------------------------------------------
// Steered conv for MI455X (gfx1250, wave32, WMMA + TDM).
// Two-stage basis-factorized formulation, fused per 8x8 pixel tile:
//   halo:    TENSOR_LOAD_TO_LDS (interior tiles) / manual path (borders)
//   Stage A: per cin, y[22 basis][px] = Abasis(32x64) x patch(64x16)   (WMMA)
//   Stage B: 9 convs,  acc[32][px]  += Amix(32x64)  x y_grp(64x16)    (WMMA)
//   Epilogue: steering nonlinearity + bias, in registers, store f32.
// ---------------------------------------------------------------------------

typedef __attribute__((ext_vector_type(16))) _Float16 v16h;
typedef __attribute__((ext_vector_type(8)))  float    v8f;
typedef unsigned int u32x4 __attribute__((ext_vector_type(4)));
typedef int          i32x8 __attribute__((ext_vector_type(8)));
typedef int          i32x4 __attribute__((ext_vector_type(4)));

#define CIN   16
#define COUT  32
#define HW    512
#define TILE  8
#define HALO  14                       // TILE + 6
#define NPX   64                       // TILE*TILE
#define KPAD  64
#define NGRP  7
#define XH_FLOATS    (CIN*HALO*HALO)   // 3136 f32
#define PATCH_BYTEOFF 12544            // 3136*4
#define PATCH_HALVES (NPX*KPAD)        // 4096
#define Y_BYTEOFF    20736             // 12544 + 8192
#define Y_HALVES     (NGRP*NPX*KPAD)   // 28672
#define SCR_BYTEOFF  78080             // 20736 + 57344
#define SMEM_BYTES   78112
#define ABASIS_HALVES (32*KPAD)        // 2048
#define AMIX_OFF     ABASIS_HALVES
#define WS_HALVES    (ABASIS_HALVES + 9*32*KPAD)  // 20480 halves = 40 KB

// ---- WMMA helper -----------------------------------------------------------
__device__ __forceinline__ v8f wmma_f32_f16(v16h a, v16h b, v8f c) {
  return __builtin_amdgcn_wmma_f32_16x16x32_f16(
      false, a, false, b, (short)0, c, false, false);
}

union FragCast { uint4 q[2]; v16h h; };

// A-operand (16x32, MxK) fragment from row-major [m][64] (128B rows).
// lane<16: K = {s*32+0..7, s*32+16..23};  lane>=16: K = {+8..15, +24..31}
__device__ __forceinline__ v16h load_fragA(const _Float16* row, int s, bool hi) {
  const uint4* p = (const uint4*)(row + s*32 + (hi ? 8 : 0));
  FragCast f; f.q[0] = p[0]; f.q[1] = p[2];
  return f.h;
}
// B-operand (32x16, KxN) fragment from row-major [n][64] (lane = N).
// lane<16: K = {s*32+0..7, s*32+16..23};  lane>=16: K = {+8..15, +24..31}
__device__ __forceinline__ v16h load_fragB(const _Float16* row, int s, bool hi) {
  const uint4* p = (const uint4*)(row + s*32 + (hi ? 8 : 0));
  FragCast f; f.q[0] = p[0]; f.q[1] = p[2];
  return f.h;
}

// ---- analytic steerable basis ---------------------------------------------
// basis order m: 0..3 BASE0 rings; 4..12 COS k=1..3 x ring 1..3; 13..21 SIN.
__device__ float basis_raw(int m, int yy, int xx) {
  float Y = (float)yy - 3.0f, X = (float)xx - 3.0f;
  float r = sqrtf(Y*Y + X*X);
  if (m < 4) { float d = r - (float)m; return expf(-2.0f*d*d); }   // sigma=0.5
  float phi = atan2f(Y, X);
  int t = m - 4;
  bool iscos = (t < 9);
  if (!iscos) t -= 9;
  float kk   = (float)(t/3 + 1);
  float ring = (float)(t%3 + 1);
  float d = r - ring;
  float g = expf(-2.0f*d*d);
  return g * (iscos ? cosf(kk*phi) : sinf(kk*phi));
}

// ---- prep kernel: build packed f16 A matrices in d_ws ----------------------
// ws[0..2047]      : Abasis[m 0..31][k=ty*8+tx]
// ws[2048..20479]  : Amix[conv 0..8][o 0..31][k=cin*4+slot]
// conv order: 0=u 1=v 2=f0 3..5=fc0..2 6..8=fs0..2
__global__ void steered_prep(const float* __restrict__ w,
                             const float* __restrict__ aw,
                             _Float16* __restrict__ ws) {
  int idx = blockIdx.x * blockDim.x + threadIdx.x;
  if (idx < ABASIS_HALVES) {
    int m = idx >> 6, k = idx & 63, ty = k >> 3, tx = k & 7;
    float v = 0.f;
    if (m < 22 && ty < 7 && tx < 7) {
      float nrm = 0.f;
      for (int yy = 0; yy < 7; ++yy)
        for (int xx = 0; xx < 7; ++xx) { float t = basis_raw(m, yy, xx); nrm += t*t; }
      v = basis_raw(m, ty, tx) * rsqrtf(nrm);
    }
    ws[idx] = (_Float16)v;
  } else if (idx < WS_HALVES) {
    int j = idx - ABASIS_HALVES;
    int c = j >> 11, m = (j >> 6) & 31, k = j & 63;
    int cin = k >> 2, sl = k & 3;
    float v = 0.f;
    if (c < 2)      { if (sl < 3) v = aw[(m*CIN + cin)*3 + sl]; }          // u,v
    else if (c == 2){ v = w[(m*CIN + cin)*13 + sl]; }                      // f0
    else {
      int i = (c <= 5) ? (c - 3) : (c - 6);                                // fc_i / fs_i
      if (sl < 3) v = w[(m*CIN + cin)*13 + 4 + 3*i + sl];
    }
    ws[idx] = (_Float16)v;
  }
}

// ---- main fused kernel -----------------------------------------------------
__global__ __launch_bounds__(128) void steered_main(
    const float* __restrict__ x, const float* __restrict__ bias,
    const _Float16* __restrict__ ws, float* __restrict__ out)
{
  extern __shared__ char smraw[];
  float*    xh32  = (float*)smraw;                        // [cin][14][14] f32
  _Float16* patch = (_Float16*)(smraw + PATCH_BYTEOFF);   // [px 64][k=ty*8+tx]
  _Float16* ybuf  = (_Float16*)(smraw + Y_BYTEOFF);       // [grp 7][px 64][cin*4+sl]
  _Float16* yscr  = (_Float16*)(smraw + SCR_BYTEOFF);     // dump for masked scatter

  const int tid   = threadIdx.x;
  const int wave  = tid >> 5;
  const int lane  = tid & 31;
  const int row16 = lane & 15;
  const bool hi   = (lane >= 16);
  const int px    = wave*16 + row16;        // this wave's N-tile pixel
  const int tX = blockIdx.x, tY = blockIdx.y, b = blockIdx.z;
  const int h0 = tY*TILE - 3, w0 = tX*TILE - 3;

  // zero y (unwritten slot-3 pads must be finite; matching Amix entries are 0)
  {
    uint4 zq = {0,0,0,0};
    uint4* yq = (uint4*)ybuf;
    for (int i = tid; i < Y_HALVES/8; i += 128) yq[i] = zq;
  }

  // ---- x halo load: TDM for interior tiles, manual path for borders --------
  const bool interior = (h0 >= 0) && (w0 >= 0) &&
                        (h0 + HALO <= HW) && (w0 + HALO <= HW);
  if (interior) {
    if (tid < 32) {   // one wave issues the DMA (EXEC ignored by TDM)
      unsigned long long gaddr =
          (unsigned long long)(const void*)(x + (((size_t)b*CIN)*HW + h0)*HW + w0);
      unsigned lds_off = (unsigned)(unsigned long long)(const void*)xh32;
      u32x4 g0;
      g0[0] = 1u;                                        // count=1, normal mode
      g0[1] = lds_off;                                   // LDS dest (bytes)
      g0[2] = (unsigned)(gaddr & 0xffffffffu);           // global_addr lo
      g0[3] = (unsigned)((gaddr >> 32) & 0x01ffffffu)    // global_addr hi
              | (2u << 30);                              // type = 2 (image)
      i32x8 g1;
      g1[0] = 0x00020000;            // data_size = 4B; no multicast/pad/iterate
      g1[1] = (HW << 16);            // tensor_dim0 = 512  (W)
      g1[2] = (HW << 16);            // tensor_dim1 = 512  (H)
      g1[3] = (HALO << 16);          // tile_dim0 = 14
      g1[4] = HALO | (CIN << 16);    // tile_dim1 = 14, tile_dim2 = 16
      g1[5] = HW;                    // tensor_dim0_stride = 512 (row)
      g1[6] = 0;                     // dim0_stride hi / dim1_stride lo16
      g1[7] = (HW*HW) >> 16;         // tensor_dim1_stride = 262144 (channel)
      i32x4 g2 = { CIN, 0, 0, 0 };   // tensor_dim2 = 16
      i32x4 g3 = { 0, 0, 0, 0 };
#if __clang_major__ >= 23
      i32x8 g1b = {0,0,0,0,0,0,0,0};
      __builtin_amdgcn_tensor_load_to_lds(g0, g1, g2, g3, g1b, 0);
#else
      __builtin_amdgcn_tensor_load_to_lds(g0, g1, g2, g3, 0);
#endif
      __builtin_amdgcn_s_wait_tensorcnt(0);
    }
  } else {
    for (int i = tid; i < XH_FLOATS; i += 128) {
      int c = i / (HALO*HALO), rem = i % (HALO*HALO);
      int yy = rem / HALO, xx = rem % HALO;
      int gy = h0 + yy, gx = w0 + xx;
      float v = 0.f;
      if ((unsigned)gy < HW && (unsigned)gx < HW)
        v = x[(((size_t)b*CIN + c)*HW + gy)*HW + gx];
      xh32[i] = v;
    }
  }

  v8f zf;
  #pragma unroll
  for (int i = 0; i < 8; ++i) zf[i] = 0.f;

  v8f acc[9][2];                     // 9 convs x 2 M-tiles, live to epilogue
  #pragma unroll
  for (int c = 0; c < 9; ++c) { acc[c][0] = zf; acc[c][1] = zf; }

  __syncthreads();

  const int gmap[9] = {1,4,0,1,2,3,4,5,6};   // conv -> y basis group

  // per-thread im2col constants (i += 128 keeps k fixed, p parity fixed)
  const int kk = tid & 63;
  const int kty = kk >> 3, ktx = kk & 7;
  const bool kvalid = (kty < 7) && (ktx < 7);
  const int p0 = tid >> 6;                   // 0 or 1

  for (int cin = 0; cin < CIN; ++cin) {
    // im2col patch (f32 LDS -> f16 LDS), immediate-offset unrolled loop
    {
      const float* src = xh32 + cin*(HALO*HALO) + kty*HALO + ktx;
      _Float16* dst = patch + kk;
      #pragma unroll
      for (int pr = 0; pr < 8; ++pr) {
        #pragma unroll
        for (int j = 0; j < 4; ++j) {
          int pc = p0 + 2*j;
          float v = kvalid ? src[pr*HALO + pc] : 0.f;
          dst[(pr*8 + pc)*KPAD] = (_Float16)v;
        }
      }
    }
    __syncthreads();

    // Stage A: y[22 basis][16 px] = Abasis x patch
    v8f ya0 = zf, ya1 = zf;
    #pragma unroll
    for (int s = 0; s < 2; ++s) {
      v16h bf = load_fragB(patch + px*KPAD, s, hi);
      v16h a0 = load_fragA(ws + (      row16)*KPAD, s, hi);
      v16h a1 = load_fragA(ws + (16 +  row16)*KPAD, s, hi);
      ya0 = wmma_f32_f16(a0, bf, ya0);
      ya1 = wmma_f32_f16(a1, bf, ya1);
    }

    // scatter basis responses into grouped-K layout (predicated, branch-free)
    #pragma unroll
    for (int mt = 0; mt < 2; ++mt) {
      #pragma unroll
      for (int r = 0; r < 8; ++r) {
        int m = mt*16 + r + (hi ? 8 : 0);
        int g  = (m < 4) ? 0 : 1 + (m-4)/3;
        int sl = (m < 4) ? m : (m-4)%3;
        float v = (mt == 0) ? ya0[r] : ya1[r];
        _Float16* dst = (m < 22)
            ? &ybuf[(g*NPX + px)*KPAD + cin*4 + sl]
            : &yscr[lane & 15];
        *dst = (_Float16)v;
      }
    }
    __syncthreads();
  }

  // Stage B: 9 mixing GEMMs sharing y fragments (A streamed from L2)
  #pragma unroll
  for (int s = 0; s < 2; ++s) {
    v16h bf[NGRP];
    #pragma unroll
    for (int g = 0; g < NGRP; ++g)
      bf[g] = load_fragB(ybuf + (g*NPX + px)*KPAD, s, hi);
    #pragma unroll
    for (int mt = 0; mt < 2; ++mt) {
      #pragma unroll
      for (int c = 0; c < 9; ++c) {
        v16h af = load_fragA(ws + AMIX_OFF + (c*32 + mt*16 + row16)*KPAD, s, hi);
        acc[c][mt] = wmma_f32_f16(af, bf[gmap[c]], acc[c][mt]);
      }
    }
  }

  // Steering epilogue, all in registers
  const int pr = px >> 3, pc = px & 7;
  const int hh = tY*TILE + pr, ww = tX*TILE + pc;
  #pragma unroll
  for (int mt = 0; mt < 2; ++mt) {
    #pragma unroll
    for (int r = 0; r < 8; ++r) {
      int o = mt*16 + r + (hi ? 8 : 0);
      float u = acc[0][mt][r], v = acc[1][mt][r];
      float inv = rsqrtf(u*u + v*v + 1e-8f);
      float c1 = u*inv, s1 = v*inv;
      float c2 = c1*c1 - s1*s1, s2 = 2.f*c1*s1;
      float c3 = c2*c1 - s2*s1, s3 = s2*c1 + c2*s1;
      float val = acc[2][mt][r]
                + c1*acc[3][mt][r] + c2*acc[4][mt][r] + c3*acc[5][mt][r]
                + s1*acc[6][mt][r] + s2*acc[7][mt][r] + s3*acc[8][mt][r]
                + bias[o];
      out[(((size_t)b*COUT + o)*HW + hh)*HW + ww] = val;
    }
  }
}

// ---------------------------------------------------------------------------
extern "C" void kernel_launch(void* const* d_in, const int* in_sizes, int n_in,
                              void* d_out, int out_size, void* d_ws, size_t ws_size,
                              hipStream_t stream) {
  (void)in_sizes; (void)n_in; (void)out_size;
  if (ws_size < (size_t)WS_HALVES * sizeof(_Float16)) return;

  const float* x    = (const float*)d_in[0];
  const float* w    = (const float*)d_in[1];
  const float* aw   = (const float*)d_in[2];
  const float* bias = (const float*)d_in[3];
  _Float16* ws = (_Float16*)d_ws;
  float* out = (float*)d_out;

  // 1) pack basis + mixing matrices (f16) into workspace
  steered_prep<<<(WS_HALVES + 255)/256, 256, 0, stream>>>(w, aw, ws);

  // 2) fused two-stage WMMA conv + steering epilogue
  (void)hipFuncSetAttribute((const void*)steered_main,
                            hipFuncAttributeMaxDynamicSharedMemorySize,
                            (int)SMEM_BYTES);
  dim3 grid(HW/TILE, HW/TILE, 4);
  steered_main<<<grid, 128, SMEM_BYTES, stream>>>(x, bias, ws, out);
}
